// HaloAttention_23115513987245
// MI455X (gfx1250) — compile-verified, matrix operand order checked by reading, not run on hardware
//
#include <hip/hip_runtime.h>
#include <hip/hip_bf16.h>

typedef __attribute__((ext_vector_type(16))) _Float16 v16h;
typedef __attribute__((ext_vector_type(8)))  float    v8f;
typedef __attribute__((ext_vector_type(4)))  float    v4f;

#define CDIM 512
#define NBLK 200          // b * nb = 8 * 25
#define NEGF (-3.402823466e38f)

// ---------------------------------------------------------------------------
// Async global -> LDS 16B copy (CDNA5 GLOBAL_LOAD_ASYNC_TO_LDS_B128, GV mode).
// LDS address = low 32 bits of generic pointer (flat LDS aperture truncates
// to addr[31:0]).  Tracked by ASYNCcnt; consume after cp_async_wait + barrier.
// ---------------------------------------------------------------------------
__device__ __forceinline__ void cp_async16(void* lds, const void* g) {
  unsigned l = (unsigned)(unsigned long long)lds;
  asm volatile("global_load_async_to_lds_b128 %0, %1, off" :: "v"(l), "v"(g) : "memory");
}
__device__ __forceinline__ void cp_async_wait() {
  asm volatile("s_wait_asynccnt 0x0" ::: "memory");
}

// ---------------------------------------------------------------------------
// Fragment builder: 16x32 (MxK) A-fragment or 32x16 (KxN) B-fragment from an
// LDS tile stored row-contiguous along K.  Per ISA 7.12.2: lanes 0-15 hold
// K 0..7 / 16..23, lanes 16-31 hold K 8..15 / 24..31.  Two 16B loads per lane.
// ---------------------------------------------------------------------------
__device__ __forceinline__ v16h ld_frag(const _Float16* tile, int rc0, int stride,
                                        int k0, int lane) {
  const _Float16* p = tile + (rc0 + (lane & 15)) * stride + k0 + ((lane & 16) >> 1);
  v16h f;
  *reinterpret_cast<v4f*>(&f)       = *reinterpret_cast<const v4f*>(p);
  *(reinterpret_cast<v4f*>(&f) + 1) = *reinterpret_cast<const v4f*>(p + 16);
  return f;
}

__device__ __forceinline__ v8f wmma16(v16h a, v16h b, v8f c) {
  return __builtin_amdgcn_wmma_f32_16x16x32_f16(false, a, false, b, (short)0, c,
                                                false, false);
}

// ---------------------------------------------------------------------------
// K1: f32 -> f16 weight / rel-table conversion (rel padded 31 -> 32 rows)
// ---------------------------------------------------------------------------
__global__ void prep_kernel(const float* __restrict__ Wq, const float* __restrict__ Wkv,
                            const float* __restrict__ Wout, const float* __restrict__ relh,
                            const float* __restrict__ relw,
                            _Float16* Wq_h, _Float16* Wkv_h, _Float16* Wout_h,
                            _Float16* Rw_h, _Float16* Rh_h) {
  long i = (long)blockIdx.x * 256 + threadIdx.x;
  if (i < 262144)  Wq_h[i]   = (_Float16)Wq[i];
  if (i < 524288)  Wkv_h[i]  = (_Float16)Wkv[i];
  if (i < 262144)  Wout_h[i] = (_Float16)Wout[i];
  if (i < 1024) {
    int r = (int)(i >> 5), d = (int)(i & 31);
    Rw_h[i] = (_Float16)(r < 31 ? relw[r * 32 + d] : 0.f);
    Rh_h[i] = (_Float16)(r < 31 ? relh[r * 32 + d] : 0.f);
  }
}

// ---------------------------------------------------------------------------
// K2: Q projection.  grid = 200 blocks * 8 col-tiles, 256 thr (8 waves).
// C(64x64) = q_inp(64x512) @ Wq^T.  Output scaled, laid out [blk,head,row,dh].
// ---------------------------------------------------------------------------
__global__ __launch_bounds__(256) void qproj_kernel(const float* __restrict__ x,
                                                    const _Float16* __restrict__ Wq_h,
                                                    _Float16* __restrict__ q_h) {
  __shared__ _Float16 As[64 * 40];
  __shared__ _Float16 Bs[64 * 40];
  const int bid = blockIdx.x;
  const int n = bid >> 3, obase = (bid & 7) * 64;
  const int bi = n / 25, rem = n % 25, by = rem / 5, bx = rem % 5;
  const int t = threadIdx.x, lane = t & 31, w = t >> 5;
  const int mt = (w >> 1) * 16, nh = (w & 1) * 32;
  v8f acc0 = {}, acc1 = {};
  const int cidx = t >> 3, qy = t & 7;
  const long xrow = (((long)bi * 512) * 40 + by * 8 + qy) * 40 + bx * 8;
  const int nnB = t >> 2, qB = (t & 3) * 8;

  for (int kb = 0; kb < 512; kb += 32) {
    // async copy of B weight tile straight into LDS (overlaps A conversion)
    const _Float16* bp = Wq_h + (long)(obase + nnB) * 512 + kb + qB;
    cp_async16(&Bs[nnB * 40 + qB], bp);
    if (kb + 32 < 512) __builtin_prefetch(bp + 32, 0, 1);

    const float* ap = x + xrow + (long)(kb + cidx) * 1600;
    v4f a0 = *reinterpret_cast<const v4f*>(ap);
    v4f a1 = *reinterpret_cast<const v4f*>(ap + 4);
    _Float16 av[8] = {(_Float16)a0.x, (_Float16)a0.y, (_Float16)a0.z, (_Float16)a0.w,
                      (_Float16)a1.x, (_Float16)a1.y, (_Float16)a1.z, (_Float16)a1.w};
#pragma unroll
    for (int i = 0; i < 8; ++i) As[(qy * 8 + i) * 40 + cidx] = av[i];

    cp_async_wait();
    __syncthreads();
    v16h a  = ld_frag(As, mt, 40, 0, lane);
    v16h b0 = ld_frag(Bs, nh, 40, 0, lane);
    v16h b1 = ld_frag(Bs, nh + 16, 40, 0, lane);
    acc0 = wmma16(a, b0, acc0);
    acc1 = wmma16(a, b1, acc1);
    __syncthreads();
  }
  const float scale = 0.17677669529663687f;  // 32^-0.5
  const int cl = lane & 15, rl = (lane & 16) >> 1;
#pragma unroll
  for (int tt = 0; tt < 2; ++tt) {
    v8f acc = tt ? acc1 : acc0;
    int o = obase + nh + tt * 16 + cl;
    int head = o >> 5, d = o & 31;
#pragma unroll
    for (int g = 0; g < 8; ++g) {
      int row = mt + rl + g;
      q_h[(((long)n * 16 + head) * 64 + row) * 32 + d] = (_Float16)(acc[g] * scale);
    }
  }
}

// ---------------------------------------------------------------------------
// K3: KV projection with fused halo gather (zero pad).
// grid = 200 * 4 row-tiles * 16 col-tiles.  k -> [blk,head,kv,dh],
// v -> transposed [blk,head,dh,kv] for direct B-fragment use in attn*V.
// ---------------------------------------------------------------------------
__global__ __launch_bounds__(256) void kvproj_kernel(const float* __restrict__ x,
                                                     const _Float16* __restrict__ Wkv_h,
                                                     _Float16* __restrict__ k_h,
                                                     _Float16* __restrict__ v_t) {
  __shared__ _Float16 As[64 * 40];
  __shared__ _Float16 Bs[64 * 40];
  const int bid = blockIdx.x;
  const int n = bid / 64, r2 = bid % 64;
  const int mblk = (r2 >> 4) * 64, obase = (r2 & 15) * 64;
  const int bi = n / 25, rem = n % 25, by = rem / 5, bx = rem % 5;
  const int t = threadIdx.x, lane = t & 31, w = t >> 5;
  const int mt = (w >> 1) * 16, nh = (w & 1) * 32;
  v8f acc0 = {}, acc1 = {};
  const int cidx = t >> 3, j = t & 7;
  const int nnB = t >> 2, qB = (t & 3) * 8;

  for (int kb = 0; kb < 512; kb += 32) {
    const _Float16* bp = Wkv_h + (long)(obase + nnB) * 512 + kb + qB;
    cp_async16(&Bs[nnB * 40 + qB], bp);
    if (kb + 32 < 512) __builtin_prefetch(bp + 32, 0, 1);

    const long cbase = ((long)bi * 512 + kb + cidx) * 1600;
#pragma unroll
    for (int i = 0; i < 8; ++i) {
      int rr = j * 8 + i;
      int r = mblk + rr;
      int ky = r >> 4, kx = r & 15;
      int py = by * 8 + ky - 4, px = bx * 8 + kx - 4;
      float v = 0.f;
      if (py >= 0 && py < 40 && px >= 0 && px < 40) v = x[cbase + py * 40 + px];
      As[rr * 40 + cidx] = (_Float16)v;
    }

    cp_async_wait();
    __syncthreads();
    v16h a  = ld_frag(As, mt, 40, 0, lane);
    v16h b0 = ld_frag(Bs, nh, 40, 0, lane);
    v16h b1 = ld_frag(Bs, nh + 16, 40, 0, lane);
    acc0 = wmma16(a, b0, acc0);
    acc1 = wmma16(a, b1, acc1);
    __syncthreads();
  }
  const int cl = lane & 15, rl = (lane & 16) >> 1;
#pragma unroll
  for (int tt = 0; tt < 2; ++tt) {
    v8f acc = tt ? acc1 : acc0;
    int o = obase + nh + tt * 16 + cl;
#pragma unroll
    for (int g = 0; g < 8; ++g) {
      int r = mblk + mt + rl + g;
      if (o < 512) {
        int head = o >> 5, d = o & 31;
        k_h[(((long)n * 16 + head) * 256 + r) * 32 + d] = (_Float16)acc[g];
      } else {
        int vo = o - 512, head = vo >> 5, d = vo & 31;
        v_t[(((long)n * 16 + head) * 32 + d) * 256 + r] = (_Float16)acc[g];
      }
    }
  }
}

// ---------------------------------------------------------------------------
// K4: attention core.  grid = 3200 (block,head), 128 thr (4 waves).
// Per wave (16 query rows): 4 WMMA rel-logits, 16 WMMA sim, softmax, 16 WMMA AV.
// All tile fills are ASYNCcnt-tracked global->LDS copies; the V-tile copy
// overlaps the softmax VALU section.
// ---------------------------------------------------------------------------
__global__ __launch_bounds__(128) void attn_kernel(const _Float16* __restrict__ q_h,
                                                   const _Float16* __restrict__ k_h,
                                                   const _Float16* __restrict__ v_t,
                                                   const _Float16* __restrict__ Rw_g,
                                                   const _Float16* __restrict__ Rh_g,
                                                   _Float16* __restrict__ ao_h) {
  __shared__ _Float16 Qs[64 * 40];    //  5120 B
  __shared__ _Float16 Ks[256 * 40];   // 20480 B (reused as Vt[32][264] later)
  __shared__ _Float16 Rw[32 * 40];    //  2560 B
  __shared__ _Float16 Rh[32 * 40];    //  2560 B
  __shared__ _Float16 Pn[64 * 264];   // 33792 B   total 64512 B
  const int bh = blockIdx.x;
  const int n = bh >> 4, head = bh & 15;
  const int rem = n % 25, by = rem / 5, bx = rem % 5;
  const int t = threadIdx.x, lane = t & 31, w = t >> 5;
  const _Float16* qg = q_h + (long)bh * 64 * 32;
  const _Float16* kg = k_h + (long)bh * 256 * 32;
  const _Float16* vg = v_t + (long)bh * 32 * 256;

  {  // Q tile (async)
    int r = t >> 1, o = (t & 1) * 16;
    cp_async16(&Qs[r * 40 + o],     qg + r * 32 + o);
    cp_async16(&Qs[r * 40 + o + 8], qg + r * 32 + o + 8);
  }
  for (int c = t; c < 512; c += 128) {  // K tile (async)
    int r = c >> 1, o = (c & 1) * 16;
    cp_async16(&Ks[r * 40 + o],     kg + r * 32 + o);
    cp_async16(&Ks[r * 40 + o + 8], kg + r * 32 + o + 8);
  }
  if (t < 64) {  // rel tables (async)
    int r = t >> 1, o = (t & 1) * 16;
    cp_async16(&Rw[r * 40 + o],     Rw_g + r * 32 + o);
    cp_async16(&Rw[r * 40 + o + 8], Rw_g + r * 32 + o + 8);
  } else {
    int t2 = t - 64, r = t2 >> 1, o = (t2 & 1) * 16;
    cp_async16(&Rh[r * 40 + o],     Rh_g + r * 32 + o);
    cp_async16(&Rh[r * 40 + o + 8], Rh_g + r * 32 + o + 8);
  }
  cp_async_wait();
  __syncthreads();

  const int rbase = w * 16;
  v16h a = ld_frag(Qs, rbase, 40, 0, lane);

  // rel logits for this wave's 16 rows: Lw/Lh = Q(16x32) @ rel^T(32x32)
  v8f lw0 = {}, lw1 = {}, lh0 = {}, lh1 = {};
  lw0 = wmma16(a, ld_frag(Rw, 0, 40, 0, lane), lw0);
  lw1 = wmma16(a, ld_frag(Rw, 16, 40, 0, lane), lw1);
  lh0 = wmma16(a, ld_frag(Rh, 0, 40, 0, lane), lh0);
  lh1 = wmma16(a, ld_frag(Rh, 16, 40, 0, lane), lh1);

  // sim: 16 column tiles, single WMMA each (K = dh = 32)
  v8f s[16];
#pragma unroll
  for (int kt = 0; kt < 16; ++kt) {
    v8f z = {};
    s[kt] = wmma16(a, ld_frag(Ks, kt * 16, 40, 0, lane), z);
  }
  __syncthreads();  // all waves done reading Ks -> safe to overlay with Vt

  _Float16* Vt = Ks;  // [32][264]; async copy overlaps softmax below
  for (int idx = t; idx < 512; idx += 128) {
    int r = idx >> 4, o = (idx & 15) * 16;
    cp_async16(&Vt[r * 264 + o],     vg + r * 256 + o);
    cp_async16(&Vt[r * 264 + o + 8], vg + r * 256 + o + 8);
  }

  // bias + mask (reference: where(valid, NEG, sim + bias))
  const int cl = lane & 15, rl = (lane & 16) >> 1;
  const int qyl = (rbase + rl) >> 3;  // row>>3 (row&7 == g)
  float bw[8];
#pragma unroll
  for (int g = 0; g < 8; ++g) {
    int jw = cl - g + 15;                       // width rel index, 8..30
    int srcl = (lane & 16) | (jw & 15);
    float b0v = __shfl(lw0[g], srcl);
    float b1v = __shfl(lw1[g], srcl);
    bw[g] = (jw < 16) ? b0v : b1v;
  }
  const int pxv = bx * 8 + cl - 4;
  const bool pxok = (pxv >= 0) && (pxv < 40);
#pragma unroll
  for (int kt = 0; kt < 16; ++kt) {
    int pyv = by * 8 + kt - 4;
    bool valid = pxok && (pyv >= 0) && (pyv < 40);
    int jh = kt + 15 - qyl;                     // height rel index, 8..30
    int srch = (lane & 16) | (jh & 15);
#pragma unroll
    for (int g = 0; g < 8; ++g) {
      float h0v = __shfl(lh0[g], srch);
      float h1v = __shfl(lh1[g], srch);
      float bhh = (jh < 16) ? h0v : h1v;
      float vs = s[kt][g] + bw[g] + bhh;
      s[kt][g] = valid ? NEGF : vs;
    }
  }

  // softmax over 256 columns (16 tiles in regs x 16 lanes per half)
#pragma unroll
  for (int g = 0; g < 8; ++g) {
    float m = s[0][g];
#pragma unroll
    for (int kt = 1; kt < 16; ++kt) m = fmaxf(m, s[kt][g]);
    m = fmaxf(m, __shfl_xor(m, 1));
    m = fmaxf(m, __shfl_xor(m, 2));
    m = fmaxf(m, __shfl_xor(m, 4));
    m = fmaxf(m, __shfl_xor(m, 8));
    float sum = 0.f;
#pragma unroll
    for (int kt = 0; kt < 16; ++kt) {
      float e = __expf(s[kt][g] - m);
      s[kt][g] = e;
      sum += e;
    }
    sum += __shfl_xor(sum, 1);
    sum += __shfl_xor(sum, 2);
    sum += __shfl_xor(sum, 4);
    sum += __shfl_xor(sum, 8);
    float inv = 1.0f / sum;
    int row = rbase + rl + g;
#pragma unroll
    for (int kt = 0; kt < 16; ++kt)
      Pn[row * 264 + kt * 16 + cl] = (_Float16)(s[kt][g] * inv);
  }
  cp_async_wait();    // Vt async copy complete
  __syncthreads();    // Pn + Vt visible

  // out = attn(16x256) @ v^T(256x32): 8 K-steps x 2 column tiles
  v8f o0 = {}, o1 = {};
#pragma unroll
  for (int ks = 0; ks < 8; ++ks) {
    v16h pa  = ld_frag(Pn, rbase, 264, ks * 32, lane);
    v16h vb0 = ld_frag(Vt, 0, 264, ks * 32, lane);
    v16h vb1 = ld_frag(Vt, 16, 264, ks * 32, lane);
    o0 = wmma16(pa, vb0, o0);
    o1 = wmma16(pa, vb1, o1);
  }
#pragma unroll
  for (int tt = 0; tt < 2; ++tt) {
    v8f acc = tt ? o1 : o0;
    int d = tt * 16 + cl;
#pragma unroll
    for (int g = 0; g < 8; ++g) {
      int row = rbase + rl + g;
      ao_h[((long)n * 64 + row) * 512 + head * 32 + d] = (_Float16)acc[g];
    }
  }
}

// ---------------------------------------------------------------------------
// K5: output projection + bias, scatter to NCHW f32.  Both tiles async.
// ---------------------------------------------------------------------------
__global__ __launch_bounds__(256) void outproj_kernel(const _Float16* __restrict__ ao_h,
                                                      const _Float16* __restrict__ Wout_h,
                                                      const float* __restrict__ bias,
                                                      float* __restrict__ out) {
  __shared__ _Float16 As[64 * 40];
  __shared__ _Float16 Bs[64 * 40];
  const int bid = blockIdx.x;
  const int n = bid >> 3, obase = (bid & 7) * 64;
  const int bi = n / 25, rem = n % 25, by = rem / 5, bx = rem % 5;
  const int t = threadIdx.x, lane = t & 31, w = t >> 5;
  const int mt = (w >> 1) * 16, nh = (w & 1) * 32;
  v8f acc0 = {}, acc1 = {};
  const int rA = t >> 2, qA = (t & 3) * 8;

  for (int kb = 0; kb < 512; kb += 32) {
    const _Float16* ap = ao_h + ((long)n * 64 + rA) * 512 + kb + qA;
    cp_async16(&As[rA * 40 + qA], ap);
    const _Float16* bp = Wout_h + (long)(obase + rA) * 512 + kb + qA;
    cp_async16(&Bs[rA * 40 + qA], bp);
    if (kb + 32 < 512) __builtin_prefetch(bp + 32, 0, 1);
    cp_async_wait();
    __syncthreads();
    v16h a  = ld_frag(As, mt, 40, 0, lane);
    v16h b0 = ld_frag(Bs, nh, 40, 0, lane);
    v16h b1 = ld_frag(Bs, nh + 16, 40, 0, lane);
    acc0 = wmma16(a, b0, acc0);
    acc1 = wmma16(a, b1, acc1);
    __syncthreads();
  }
  const int cl = lane & 15, rl = (lane & 16) >> 1;
#pragma unroll
  for (int tt = 0; tt < 2; ++tt) {
    v8f acc = tt ? acc1 : acc0;
    int o = obase + nh + tt * 16 + cl;
    float bo = bias[o];
#pragma unroll
    for (int g = 0; g < 8; ++g) {
      int row = mt + rl + g, qy = row >> 3, qx = row & 7;
      out[(((long)bi * 512 + o) * 40 + by * 8 + qy) * 40 + bx * 8 + qx] = acc[g] + bo;
    }
  }
}

// ---------------------------------------------------------------------------
extern "C" void kernel_launch(void* const* d_in, const int* in_sizes, int n_in,
                              void* d_out, int out_size, void* d_ws, size_t ws_size,
                              hipStream_t stream) {
  const float* x      = (const float*)d_in[0];
  const float* Wq     = (const float*)d_in[1];
  const float* Wkv    = (const float*)d_in[2];
  const float* Wout_w = (const float*)d_in[3];
  const float* Wout_b = (const float*)d_in[4];
  const float* rel_h  = (const float*)d_in[5];
  const float* rel_w  = (const float*)d_in[6];
  float* out = (float*)d_out;

  char* ws = (char*)d_ws;
  _Float16* Wq_h   = (_Float16*)(ws + 0);           //   512 KB
  _Float16* Wkv_h  = (_Float16*)(ws + 524288);      //  1024 KB
  _Float16* Wout_h = (_Float16*)(ws + 1572864);     //   512 KB
  _Float16* Rw_h   = (_Float16*)(ws + 2097152);     //     2 KB
  _Float16* Rh_h   = (_Float16*)(ws + 2099200);     //     2 KB
  _Float16* q_h    = (_Float16*)(ws + 2101248);     //  12.5 MB
  _Float16* k_h    = (_Float16*)(ws + 15208448);    //    50 MB
  _Float16* v_t    = (_Float16*)(ws + 67637248);    //    50 MB
  _Float16* ao_h   = (_Float16*)(ws + 120066048);   //  12.5 MB

  prep_kernel<<<dim3(2048), dim3(256), 0, stream>>>(Wq, Wkv, Wout_w, rel_h, rel_w,
                                                    Wq_h, Wkv_h, Wout_h, Rw_h, Rh_h);
  qproj_kernel<<<dim3(NBLK * 8), dim3(256), 0, stream>>>(x, Wq_h, q_h);
  kvproj_kernel<<<dim3(NBLK * 64), dim3(256), 0, stream>>>(x, Wkv_h, k_h, v_t);
  attn_kernel<<<dim3(NBLK * 16), dim3(128), 0, stream>>>(q_h, k_h, v_t, Rw_h, Rh_h, ao_h);
  outproj_kernel<<<dim3(NBLK * 8), dim3(256), 0, stream>>>(ao_h, Wout_h, Wout_b, out);
}
